// QKVAttentionLegacy_78056735637905
// MI455X (gfx1250) — matrix operand order repaired and checked by reading, not compiled, not run on hardware
//
#include <hip/hip_runtime.h>
#include <hip/hip_bf16.h>
#include <math.h>

typedef __attribute__((ext_vector_type(16))) __bf16 v16bf;
typedef __attribute__((ext_vector_type(8)))  float  v8f;

#define SEQ   2048
#define CH    64
#define TQ    128     // query rows per block (8 waves)
#define TS    64      // key rows per tile
#define NWAVE 8
#define NTHR  256

union Frag {
    v16bf v;
    unsigned short u[16];
    uint4 q[2];
};

__device__ __forceinline__ unsigned short f2bf(float f) {
    unsigned int x = __float_as_uint(f);
    x += 0x7FFFu + ((x >> 16) & 1u);      // round-to-nearest-even
    return (unsigned short)(x >> 16);
}

__device__ __forceinline__ unsigned int pack_bf2(float lo, float hi) {
#if __has_builtin(__builtin_amdgcn_cvt_pk_bf16_f32)
    auto r = __builtin_amdgcn_cvt_pk_bf16_f32(lo, hi);   // v_cvt_pk_bf16_f32
    return __builtin_bit_cast(unsigned int, r);
#else
    return (unsigned int)f2bf(lo) | ((unsigned int)f2bf(hi) << 16);
#endif
}

__device__ __forceinline__ float fastexp2(float x) {
#if __has_builtin(__builtin_amdgcn_exp2f)
    return __builtin_amdgcn_exp2f(x);     // bare v_exp_f32
#else
    return exp2f(x);
#endif
}

// sum across the 16-lane half that owns a row (xor bits 0..3 only)
__device__ __forceinline__ float redsum16(float x) {
    #pragma unroll
    for (int off = 8; off > 0; off >>= 1)
        x += __shfl_xor(x, off, 32);
    return x;
}

__global__ __launch_bounds__(NTHR)
void qkv_attn_fa_kernel(const float* __restrict__ qkv, float* __restrict__ out)
{
    const int tid  = threadIdx.x;
    const int lane = tid & 31;
    const int w    = tid >> 5;            // wave 0..7
    const int bh   = blockIdx.y;          // batch-head 0..31
    const int t0   = blockIdx.x * TQ;     // query tile base

    // ch^-0.25 * sqrt(log2(e)): folds softmax's log2(e) into the matmul so
    // exp(S) == exp2(S'), removing a v_mul before every v_exp_f32.
    const float scale = 0.42466090014469244f;

    // one raw LDS block, carved manually so the output stage can reuse it
    __shared__ __align__(16) unsigned char smem[49152];
    unsigned short* Qs = (unsigned short*)(smem);           // [TQ][CH]  16 KB
    unsigned short* Ks = (unsigned short*)(smem + 16384);   // [TS][CH]   8 KB
    unsigned short* Vt = (unsigned short*)(smem + 24576);   // [CH][TS]   8 KB
    unsigned short* Ps = (unsigned short*)(smem + 32768);   // per-wave [16][TS] 16 KB
    float*          OutS = (float*)(smem);                  // reuse: [CH][TQ] 32 KB

    const float* qbase = qkv + (size_t)bh * 192 * SEQ;      // rows [0,64)
    const float* kbase = qbase + (size_t)64  * SEQ;
    const float* vbase = qbase + (size_t)128 * SEQ;

    // Stage Q tile as bf16 pairs: LDS layout [t][c], u32 stores
    for (int i = tid; i < TQ * CH / 2; i += NTHR) {
        int r  = i & (TQ - 1);
        int c0 = (i >> 7) * 2;
        float q0 = qbase[(size_t)c0       * SEQ + t0 + r] * scale;
        float q1 = qbase[(size_t)(c0 + 1) * SEQ + t0 + r] * scale;
        *(unsigned int*)&Qs[r * CH + c0] = pack_bf2(q0, q1);
    }
    __syncthreads();

    const int half  = (lane >= 16) ? 1 : 0;
    const int m     = lane & 15;
    const int half8 = half * 8;
    unsigned short* Pw = Ps + w * 16 * TS;

    // Hoist the two Q A-fragments (loop-invariant): rows w*16 + m, K = c
    Frag qa[2];
    #pragma unroll
    for (int kk = 0; kk < 2; ++kk) {
        const unsigned short* qrow = &Qs[(w * 16 + m) * CH + kk * 32];
        qa[kk].q[0] = *(const uint4*)(qrow + half * 16);
        qa[kk].q[1] = *(const uint4*)(qrow + half * 16 + 8);
    }

    v8f o[4];
    #pragma unroll
    for (int cn = 0; cn < 4; ++cn) o[cn] = (v8f){};

    float lrun[8];                        // per-lane partial row sums
    #pragma unroll
    for (int i = 0; i < 8; ++i) lrun[i] = 0.0f;

    for (int s0 = 0; s0 < SEQ; s0 += TS) {
        __syncthreads();   // previous tile fully consumed

        // Stage K tile [s][c] (bf16 pairs over c)
        for (int i = tid; i < TS * CH / 2; i += NTHR) {
            int r  = i & 63;
            int c0 = (i >> 6) * 2;
            float k0 = kbase[(size_t)c0       * SEQ + s0 + r] * scale;
            float k1 = kbase[(size_t)(c0 + 1) * SEQ + s0 + r] * scale;
            *(unsigned int*)&Ks[r * CH + c0] = pack_bf2(k0, k1);
        }
        // Stage V tile transposed [c][s] (float2 loads, bf16 pairs over s)
        for (int i = tid; i < CH * TS / 2; i += NTHR) {
            int sp = (i & 31) * 2;
            int c  = i >> 5;
            float2 vv = *(const float2*)&vbase[(size_t)c * SEQ + s0 + sp];
            *(unsigned int*)&Vt[c * TS + sp] = pack_bf2(vv.x, vv.y);
        }
        __syncthreads();

        // Uniform prefetch of the full next tile (global_prefetch_b8):
        // 64 rows x 2 arrays x 2 cachelines, no divergent branch.
        if (s0 + TS < SEQ) {
            int r   = tid & 63;
            int sel = (tid >> 6) & 1;
            int ofs = (tid >> 7) * 32;
            const float* pf = (sel ? vbase : kbase) + (size_t)r * SEQ + s0 + TS + ofs;
            __builtin_prefetch(pf, 0, 1);
        }

        // ---- S' = (Q*scale')(K*scale')^T = S * log2(e), per-wave 16x64 ----
        v8f S[4];
        #pragma unroll
        for (int sn = 0; sn < 4; ++sn) {
            v8f acc = (v8f){};
            #pragma unroll
            for (int kk = 0; kk < 2; ++kk) {
                Frag b;   // B 32x16: K = c (contiguous), N = s = lane&15
                const unsigned short* krow = &Ks[(sn * 16 + m) * CH + kk * 32 + half * 16];
                b.q[0] = *(const uint4*)(krow);
                b.q[1] = *(const uint4*)(krow + 8);
                acc = __builtin_amdgcn_wmma_f32_16x16x32_bf16(
                          false, qa[kk].v, false, b.v, (short)0, acc, false, false);
            }
            S[sn] = acc;
        }

        // ---- unnormalized exp2 (== exp of true scores); shift-free softmax ----
        // Convert in pairs (v_cvt_pk_bf16_f32) and store halves directly
        // (low -> ds_store_b16, high -> ds_store_b16_d16_hi).
        #pragma unroll
        for (int i = 0; i < 8; ++i) {
            const float p0 = fastexp2(S[0][i]);
            const float p1 = fastexp2(S[1][i]);
            const float p2 = fastexp2(S[2][i]);
            const float p3 = fastexp2(S[3][i]);
            lrun[i] += (p0 + p1) + (p2 + p3);
            const unsigned int pk01 = pack_bf2(p0, p1);
            const unsigned int pk23 = pack_bf2(p2, p3);
            unsigned short* prow = &Pw[(i + half8) * TS + m];
            prow[ 0] = (unsigned short)(pk01);
            prow[16] = (unsigned short)(pk01 >> 16);
            prow[32] = (unsigned short)(pk23);
            prow[48] = (unsigned short)(pk23 >> 16);
        }
        // Pw written and read by the same wave; per-wave LDS ops are in-order.

        // ---- O += P * V^T ----
        #pragma unroll
        for (int kk = 0; kk < 2; ++kk) {
            const int sb = kk * 32;
            Frag a;   // A 16x32: M = t row, K = s
            const unsigned short* prow = &Pw[m * TS + sb];
            a.q[0] = *(const uint4*)(prow + half * 16);
            a.q[1] = *(const uint4*)(prow + half * 16 + 8);
            #pragma unroll
            for (int cn = 0; cn < 4; ++cn) {
                Frag b;   // B 32x16: K = s (contiguous in Vt), N = c = lane&15
                const unsigned short* vrow = &Vt[(cn * 16 + m) * TS + sb + half * 16];
                b.q[0] = *(const uint4*)(vrow);
                b.q[1] = *(const uint4*)(vrow + 8);
                o[cn] = __builtin_amdgcn_wmma_f32_16x16x32_bf16(
                            false, a.v, false, b.v, (short)0, o[cn], false, false);
            }
        }
    }

    // ---- finalize: row-sum reduction, normalize, stage to LDS, coalesced out ----
    float invl[8];
    #pragma unroll
    for (int i = 0; i < 8; ++i) invl[i] = 1.0f / redsum16(lrun[i]);

    __syncthreads();   // done with Qs/Ks/Vt — safe to reuse smem as OutS
    #pragma unroll
    for (int cn = 0; cn < 4; ++cn) {
        const int c = cn * 16 + m;
        float4 lo, hi;
        lo.x = o[cn][0] * invl[0]; lo.y = o[cn][1] * invl[1];
        lo.z = o[cn][2] * invl[2]; lo.w = o[cn][3] * invl[3];
        hi.x = o[cn][4] * invl[4]; hi.y = o[cn][5] * invl[5];
        hi.z = o[cn][6] * invl[6]; hi.w = o[cn][7] * invl[7];
        float* orow = &OutS[c * TQ + w * 16 + half8];
        *(float4*)(orow)     = lo;
        *(float4*)(orow + 4) = hi;
    }
    __syncthreads();

    float* ob = out + (size_t)bh * CH * SEQ;
    for (int idx = tid; idx < CH * TQ / 4; idx += NTHR) {
        int c  = idx >> 5;              // 32 float4 per row of TQ
        int t4 = (idx & 31) * 4;
        *(float4*)&ob[(size_t)c * SEQ + t0 + t4] = *(const float4*)&OutS[c * TQ + t4];
    }
}

extern "C" void kernel_launch(void* const* d_in, const int* in_sizes, int n_in,
                              void* d_out, int out_size, void* d_ws, size_t ws_size,
                              hipStream_t stream) {
    const float* qkv = (const float*)d_in[0];
    float* out = (float*)d_out;
    dim3 grid(SEQ / TQ, 32);   // 16 query tiles x 32 batch-heads
    dim3 block(NTHR);          // 8 wave32
    qkv_attn_fa_kernel<<<grid, block, 0, stream>>>(qkv, out);
}
